// DINAttention_63909113365121
// MI455X (gfx1250) — matrix operand order repaired and checked by reading, not compiled, not run on hardware
//
#include <hip/hip_runtime.h>
#include <hip/hip_bf16.h>

// ---------------------------------------------------------------------------
// DIN attention, MI455X (gfx1250, wave32, WMMA bf16 -> f32).
//   h = cand@(W1a+W1c)  +  [clicks | cand*clicks] @ [[W1b-W1c];[W1d]]  + b1
// GEMM per batch: M=112 (pad of 100), N=64, K=128, bf16 WMMA 16x16x32.
// ---------------------------------------------------------------------------

typedef __attribute__((ext_vector_type(16))) __bf16 v16bf;
typedef __attribute__((ext_vector_type(8)))  float  v8f;

constexpr int BB   = 4096;   // batch
constexpr int LL   = 100;    // clicks per batch
constexpr int DD   = 64;     // feature dim / hidden dim
constexpr int LPAD = 112;    // M padded to 7*16
constexpr int KK   = 128;    // reduced K
constexpr int SA   = 136;    // LDS row stride (elems) for A  (272B -> bank skew 4)
constexpr int SB   = 136;    // LDS row stride (elems) for B^T
constexpr int NPADROWS = LPAD - LL;   // 12 zero rows -> h = hc[n] exactly

// A fragment: lane (M = lane&15), K-half = (lane>>4)*8; elems 0-7 = K kb..kb+7,
// elems 8-15 = K kb+16..kb+23  (ISA 7.12.2, 16-bit A 16x32).
__device__ __forceinline__ v16bf frag_a(const __bf16* rowp, int kb) {
  union { uint4 u[2]; v16bf v; } un;
  un.u[0] = *reinterpret_cast<const uint4*>(rowp + kb);
  un.u[1] = *reinterpret_cast<const uint4*>(rowp + kb + 16);
  return un.v;
}
// B fragment (B^T in LDS): lane (N = lane&15), K-half = (lane>>4)*8; elems 0-7 =
// K kb..kb+7, elems 8-15 = K kb+16..kb+23 (same K order as A).
__device__ __forceinline__ v16bf frag_b(const __bf16* colp, int kb) {
  union { uint4 u[2]; v16bf v; } un;
  un.u[0] = *reinterpret_cast<const uint4*>(colp + kb);
  un.u[1] = *reinterpret_cast<const uint4*>(colp + kb + 16);
  return un.v;
}

// ---------------- Kernel 1: GEMM + per-block BN partial sums ----------------
__global__ __launch_bounds__(256) void din_stats_kernel(
    const float* __restrict__ cand, const float* __restrict__ clicks,
    const float* __restrict__ W1,   const float* __restrict__ b1,
    float* __restrict__ psum, float* __restrict__ psq)
{
  __shared__ __align__(16) __bf16 Abf[LPAD * SA];
  __shared__ __align__(16) __bf16 Btr[DD * SB];
  __shared__ float candf[DD];
  __shared__ float hcp[4][DD];
  __shared__ float hc[DD];
  __shared__ float tsum[28 * 16];
  __shared__ float tsq [28 * 16];

  const int b = blockIdx.x, tid = threadIdx.x;
  const int lane = tid & 31, wv = tid >> 5;

  if (tid < DD) candf[tid] = cand[b * DD + tid];
  __syncthreads();

  // A = [clicks | cand*clicks] as bf16
  for (int i = tid; i < LL * DD; i += 256) {
    float v = clicks[(size_t)b * LL * DD + i];
    int l = i >> 6, k = i & 63;
    Abf[l * SA + k]      = (__bf16)v;
    Abf[l * SA + 64 + k] = (__bf16)(v * candf[k]);
  }
  for (int i = tid; i < NPADROWS * KK; i += 256) {          // zero pad rows
    int l = LL + (i >> 7), k = i & 127;
    Abf[l * SA + k] = (__bf16)0.f;
  }
  // B^T: rows n, cols k. k<64: W1b-W1c ; k>=64: W1d
  for (int i = tid; i < DD * KK; i += 256) {
    int n = i >> 7, k = i & 127;
    float v = (k < 64) ? (W1[(64 + k) * DD + n] - W1[(128 + k) * DD + n])
                       : W1[(192 + k - 64) * DD + n];
    Btr[n * SB + k] = (__bf16)v;
  }
  // hc[n] = b1 + cand @ (W1a + W1c), fp32
  {
    int n = tid & 63, g = tid >> 6;
    float s = 0.f;
    for (int k = g * 16; k < g * 16 + 16; ++k)
      s += candf[k] * (W1[k * DD + n] + W1[(128 + k) * DD + n]);
    hcp[g][n] = s;
  }
  __syncthreads();
  if (tid < DD) hc[tid] = b1[tid] + hcp[0][tid] + hcp[1][tid] + hcp[2][tid] + hcp[3][tid];
  __syncthreads();

  const int rc  = lane & 15;
  const int ka  = (lane >> 4) << 3;
  const int kb2 = (lane >> 4) << 4;
  for (int t = wv; t < 28; t += 8) {                // wave-uniform: EXEC all-1
    int mt = t >> 2, nt = t & 3;
    const __bf16* ap = &Abf[(mt * 16 + rc) * SA];
    const __bf16* bp = &Btr[(nt * 16 + rc) * SB];
    // prefetch all fragments -> single dscnt wait, back-to-back WMMAs
    v16bf af[4], bf_[4];
#pragma unroll
    for (int kt = 0; kt < 4; ++kt) {
      af[kt]  = frag_a(ap, kt * 32 + ka);
      bf_[kt] = frag_b(bp, kt * 32 + ka);
    }
    v8f acc = {};
#pragma unroll
    for (int kt = 0; kt < 4; ++kt)
      acc = __builtin_amdgcn_wmma_f32_16x16x32_bf16(false, af[kt], false, bf_[kt],
                                                    (short)0, acc, false, false);
    // Unconditional accumulation: padded rows contribute exactly hc[n] each
    // (A rows zeroed); corrected in closed form in the final reduction.
    int n = nt * 16 + rc;
    float bias = hc[n];
    float s = 0.f, q = 0.f;
#pragma unroll
    for (int r = 0; r < 8; ++r) { float h = acc[r] + bias; s += h; q += h * h; }
    s += __shfl_xor(s, 16, 32);                     // merge the two M-halves (rows 0-7 / 8-15)
    q += __shfl_xor(q, 16, 32);
    if (lane < 16) { tsum[t * 16 + lane] = s; tsq[t * 16 + lane] = q; }
  }
  __syncthreads();
  if (tid < DD) {                                   // fixed-order (deterministic)
    int nt = tid >> 4, r = tid & 15;
    float s = 0.f, q = 0.f;
#pragma unroll
    for (int mt = 0; mt < 7; ++mt) {
      s += tsum[(mt * 4 + nt) * 16 + r];
      q += tsq [(mt * 4 + nt) * 16 + r];
    }
    float hcn = hc[tid];
    s -= (float)NPADROWS * hcn;                     // remove 12 pad rows
    q -= (float)NPADROWS * hcn * hcn;
    psum[(size_t)tid * BB + b] = s;
    psq [(size_t)tid * BB + b] = q;
  }
}

// ---------------- Kernel 2: finalize BN stats (deterministic tree) ----------
__global__ __launch_bounds__(256) void din_finalize_kernel(
    const float* __restrict__ psum, const float* __restrict__ psq,
    float* __restrict__ mu_out, float* __restrict__ rs_out)
{
  __shared__ float r1[256], r2[256];
  const int n = blockIdx.x, tid = threadIdx.x;
  float s = 0.f, q = 0.f;
  for (int i = tid; i < BB; i += 256) {
    s += psum[(size_t)n * BB + i];
    q += psq [(size_t)n * BB + i];
  }
  r1[tid] = s; r2[tid] = q;
  __syncthreads();
  for (int off = 128; off > 0; off >>= 1) {
    if (tid < off) { r1[tid] += r1[tid + off]; r2[tid] += r2[tid + off]; }
    __syncthreads();
  }
  if (tid == 0) {
    const float inv = 1.f / 409600.f;
    float mu  = r1[0] * inv;
    float var = r2[0] * inv - mu * mu;
    mu_out[n] = mu;
    rs_out[n] = rsqrtf(fmaxf(var, 0.f) + 1e-9f);
  }
}

// ---------------- Kernel 3: GEMM + Dice + softmax + weighted sum ------------
__global__ __launch_bounds__(256) void din_fused_kernel(
    const float* __restrict__ cand, const float* __restrict__ clicks,
    const int*   __restrict__ mask, const float* __restrict__ W1,
    const float* __restrict__ b1,   const float* __restrict__ alpha,
    const float* __restrict__ W2,   const float* __restrict__ b2,
    const float* __restrict__ ws_mu, const float* __restrict__ ws_rs,
    float* __restrict__ out)
{
  __shared__ __align__(16) __bf16 Abf[LPAD * SA];
  __shared__ __align__(16) __bf16 Btr[DD * SB];
  __shared__ float clicksf[LL * DD];
  __shared__ float candf[DD];
  __shared__ float hcp[4][DD];
  __shared__ float hc[DD];
  __shared__ float h_lds[LPAD * DD];
  __shared__ float muS[DD], rsS[DD], alS[DD], w2S[DD];
  __shared__ float wred[LPAD];
  __shared__ float pw[LL];
  __shared__ float pc[4][DD];

  const int b = blockIdx.x, tid = threadIdx.x;
  const int lane = tid & 31, wv = tid >> 5;

  if (tid < DD) candf[tid] = cand[b * DD + tid];
  if (tid >= 64 && tid < 128) {
    int n = tid - 64;
    muS[n] = ws_mu[n]; rsS[n] = ws_rs[n]; alS[n] = alpha[n]; w2S[n] = W2[n];
  }
  __syncthreads();

  for (int i = tid; i < LL * DD; i += 256) {
    float v = clicks[(size_t)b * LL * DD + i];
    clicksf[i] = v;                                   // fp32 copy for output
    int l = i >> 6, k = i & 63;
    Abf[l * SA + k]      = (__bf16)v;
    Abf[l * SA + 64 + k] = (__bf16)(v * candf[k]);
  }
  for (int i = tid; i < NPADROWS * KK; i += 256) {
    int l = LL + (i >> 7), k = i & 127;
    Abf[l * SA + k] = (__bf16)0.f;
  }
  for (int i = tid; i < DD * KK; i += 256) {
    int n = i >> 7, k = i & 127;
    float v = (k < 64) ? (W1[(64 + k) * DD + n] - W1[(128 + k) * DD + n])
                       : W1[(192 + k - 64) * DD + n];
    Btr[n * SB + k] = (__bf16)v;
  }
  {
    int n = tid & 63, g = tid >> 6;
    float s = 0.f;
    for (int k = g * 16; k < g * 16 + 16; ++k)
      s += candf[k] * (W1[k * DD + n] + W1[(128 + k) * DD + n]);
    hcp[g][n] = s;
  }
  __syncthreads();
  if (tid < DD) hc[tid] = b1[tid] + hcp[0][tid] + hcp[1][tid] + hcp[2][tid] + hcp[3][tid];
  __syncthreads();

  // ---- WMMA GEMM: h_lds[l][n] = (A@B)[l][n] + hc[n] ----
  const int rc  = lane & 15;
  const int ka  = (lane >> 4) << 3;
  const int kb2 = (lane >> 4) << 4;
  for (int t = wv; t < 28; t += 8) {
    int mt = t >> 2, nt = t & 3;
    const __bf16* ap = &Abf[(mt * 16 + rc) * SA];
    const __bf16* bp = &Btr[(nt * 16 + rc) * SB];
    v16bf af[4], bf_[4];
#pragma unroll
    for (int kt = 0; kt < 4; ++kt) {
      af[kt]  = frag_a(ap, kt * 32 + ka);
      bf_[kt] = frag_b(bp, kt * 32 + ka);
    }
    v8f acc = {};
#pragma unroll
    for (int kt = 0; kt < 4; ++kt)
      acc = __builtin_amdgcn_wmma_f32_16x16x32_bf16(false, af[kt], false, bf_[kt],
                                                    (short)0, acc, false, false);
    int n     = nt * 16 + rc;
    int mrow0 = mt * 16 + ((lane >> 4) << 3);
    float bias = hc[n];
#pragma unroll
    for (int r = 0; r < 8; ++r)
      h_lds[(mrow0 + r) * DD + n] = acc[r] + bias;
  }
  const float b2v = b2[0];
  __syncthreads();

  // ---- Dice + w = d @ W2 + b2 : one wave per row, shuffle reduce ----
  for (int l = wv; l < LL; l += 8) {
    float s = 0.f;
#pragma unroll
    for (int h2 = 0; h2 < 2; ++h2) {
      int n = lane + h2 * 32;
      float h  = h_lds[l * DD + n];
      float hn = (h - muS[n]) * rsS[n];
      float p  = 1.f / (1.f + __expf(-hn));
      float d  = (p + (1.f - p) * alS[n]) * h;
      s += d * w2S[n];
    }
#pragma unroll
    for (int off = 16; off > 0; off >>= 1) s += __shfl_xor(s, off, 32);
    if (lane == 0) wred[l] = s + b2v;
  }
  __syncthreads();

  // ---- masked softmax over L (wave 0) ----
  if (wv == 0) {
    float v[4];
#pragma unroll
    for (int j = 0; j < 4; ++j) {
      int l = lane + 32 * j;
      v[j] = (l < LL && mask[(size_t)b * LL + l]) ? wred[l] : -1e9f;
    }
    float mx = fmaxf(fmaxf(v[0], v[1]), fmaxf(v[2], v[3]));
#pragma unroll
    for (int off = 16; off > 0; off >>= 1) mx = fmaxf(mx, __shfl_xor(mx, off, 32));
    float e[4]; float s = 0.f;
#pragma unroll
    for (int j = 0; j < 4; ++j) {
      int l = lane + 32 * j;
      e[j] = (l < LL) ? __expf(v[j] - mx) : 0.f;
      s += e[j];
    }
#pragma unroll
    for (int off = 16; off > 0; off >>= 1) s += __shfl_xor(s, off, 32);
    float inv = 1.f / s;
#pragma unroll
    for (int j = 0; j < 4; ++j) {
      int l = lane + 32 * j;
      if (l < LL) pw[l] = e[j] * inv;
    }
  }
  __syncthreads();

  // ---- out[b][n] = sum_l pw[l] * clicks[b][l][n] (fp32) ----
  {
    int n = tid & 63, g = tid >> 6;
    float s = 0.f;
    for (int l = g * 25; l < g * 25 + 25; ++l)
      s += pw[l] * clicksf[l * DD + n];
    pc[g][n] = s;
  }
  __syncthreads();
  if (tid < DD)
    out[(size_t)b * DD + tid] = pc[0][tid] + pc[1][tid] + pc[2][tid] + pc[3][tid];
}

// ---------------------------------------------------------------------------
extern "C" void kernel_launch(void* const* d_in, const int* in_sizes, int n_in,
                              void* d_out, int out_size, void* d_ws, size_t ws_size,
                              hipStream_t stream) {
  const float* cand   = (const float*)d_in[0];
  const float* clicks = (const float*)d_in[1];
  const int*   mask   = (const int*)  d_in[2];
  const float* W1     = (const float*)d_in[3];
  const float* b1     = (const float*)d_in[4];
  const float* alpha  = (const float*)d_in[5];
  const float* W2     = (const float*)d_in[6];
  const float* b2     = (const float*)d_in[7];
  float* out = (float*)d_out;

  // ws layout (floats): [0,64) mu | [64,128) rsig | psum 64*4096 | psq 64*4096
  float* wsf  = (float*)d_ws;
  float* mu   = wsf;
  float* rs   = wsf + 64;
  float* psum = wsf + 128;
  float* psq  = psum + (size_t)DD * BB;

  din_stats_kernel<<<BB, 256, 0, stream>>>(cand, clicks, W1, b1, psum, psq);
  din_finalize_kernel<<<DD, 256, 0, stream>>>(psum, psq, mu, rs);
  din_fused_kernel<<<BB, 256, 0, stream>>>(cand, clicks, mask, W1, b1, alpha,
                                           W2, b2, mu, rs, out);
}
